// InfoGCN_70136815943863
// MI455X (gfx1250) — compile-verified, matrix operand order checked by reading, not run on hardware
//
#include <hip/hip_runtime.h>

#define N_S 64
#define C_C 128
#define T_T 64
#define V_V 25
#define TV  (T_T * V_V)

typedef __attribute__((ext_vector_type(16))) __bf16 bf16x16;
typedef __attribute__((ext_vector_type(2)))  __bf16 bf16x2;
typedef __attribute__((ext_vector_type(2)))  float  f32x2;
typedef __attribute__((ext_vector_type(8)))  float  f32x8;

struct alignas(16) Smem {
  unsigned short W[4][C_C * C_C];   // bf16 weights, row-major [o][c]
  unsigned short Ab[32 * 32];       // bf16 A zero-padded to 32x32: Ab[v][u] = A[v][u]
  float          bias[4][C_C];
  unsigned short xb[C_C * 32];      // bf16 x, row-major [c][v], pitch 32 (WMMA A-source)
  unsigned short ybT[32 * C_C];     // bf16 y, transposed [v][c]      (WMMA B-source)
  unsigned short yb2[32 * C_C];     // bf16 x3, transposed [v][c] for final projection
  float          zt[C_C * V_V];     // z_t
  float          zc[C_C * V_V];     // z_cand
  float          pec[C_C];          // pe[:, 3t]
  float          hstage[C_C * V_V]; // async-staged h[:, :, t, :]
  float          ring[6][C_C * V_V];// prev_f ring (f32)
};

__device__ __forceinline__ unsigned short f2bf(float x) {
  union { __bf16 b; unsigned short u; } cv;
  cv.b = (__bf16)x;                 // native fptrunc -> v_cvt_pk_bf16_f32
  return cv.u;
}

// Pack two f32 into a bf16 pair via vector fptrunc -> single v_cvt_pk_bf16_f32.
__device__ __forceinline__ unsigned pkbf(float a, float b) {
  f32x2 t = {a, b};
  union { bf16x2 v; unsigned u; } cv;
  cv.v = __builtin_convertvector(t, bf16x2);
  return cv.u;
}

__device__ __forceinline__ float fast_tanh(float x) {
#if __has_builtin(__builtin_amdgcn_tanhf)
  return __builtin_amdgcn_tanhf(x); // CDNA5 v_tanh_f32
#else
  return tanhf(x);
#endif
}

__device__ __forceinline__ void wait_async0() {
#if __has_builtin(__builtin_amdgcn_s_wait_asynccnt)
  __builtin_amdgcn_s_wait_asynccnt(0);
#else
  asm volatile("s_wait_asynccnt 0x0" ::: "memory");
#endif
}

// Async global->LDS copy of one h time-slice (3200 f32), ASYNCcnt-tracked.
__device__ __forceinline__ void issue_hslice_async(Smem& s, const float* hn,
                                                   int t, int tid) {
  for (int idx = tid; idx < C_C * V_V; idx += 256) {
    int c = idx / V_V, v = idx - c * V_V;
    const float* gp = hn + (size_t)c * TV + t * V_V + v;
    unsigned long long ga = (unsigned long long)(const void*)gp;
    unsigned lds = (unsigned)(unsigned long long)(const void*)&s.hstage[idx];
    asm volatile("global_load_async_to_lds_b32 %0, %1, off"
                 :: "v"(lds), "v"(ga) : "memory");
  }
}

// Load one 16x32 bf16 WMMA fragment; each VGPR is one aligned b32 LDS load.
__device__ __forceinline__ bf16x16 ld_frag(const unsigned short* base, int rc,
                                           int pitch, int kofs, int laneHi) {
  const unsigned int* p =
      (const unsigned int*)(base + rc * pitch + kofs) + laneHi * 4;
  union { bf16x16 v; unsigned int u[8]; } f;
  f.u[0] = p[0]; f.u[1] = p[1]; f.u[2] = p[2];  f.u[3] = p[3];
  f.u[4] = p[8]; f.u[5] = p[9]; f.u[6] = p[10]; f.u[7] = p[11];
  return f.v;
}

__device__ __forceinline__ f32x8 wmma_bf16(bf16x16 a, bf16x16 b, f32x8 c) {
  return __builtin_amdgcn_wmma_f32_16x16x32_bf16(false, a, false, b, (short)0,
                                                 c, false, false);
}

__device__ __forceinline__ void store8(unsigned short* dst, const float v[8]) {
  uint4 p;
  p.x = pkbf(v[0], v[1]);
  p.y = pkbf(v[2], v[3]);
  p.z = pkbf(v[4], v[5]);
  p.w = pkbf(v[6], v[7]);
  *(uint4*)dst = p;
}

// Graph conv: ybT[v][c] = sum_u xb[c][u] * Ab[v][u]   (M=128, N=32, K=32)
__device__ __forceinline__ void gemm_graph(Smem& s, int m0, int laneLo, int laneHi) {
  bf16x16 a  = ld_frag(s.xb, m0 + laneLo, 32, 0, laneHi);
  bf16x16 b0 = ld_frag(s.Ab, laneLo,      32, 0, laneHi);
  bf16x16 b1 = ld_frag(s.Ab, 16 + laneLo, 32, 0, laneHi);
  f32x8 z = {0, 0, 0, 0, 0, 0, 0, 0};
  f32x8 acc0 = wmma_bf16(a, b0, z);
  f32x8 acc1 = wmma_bf16(a, b1, z);
  float t0[8], t1[8];
#pragma unroll
  for (int r = 0; r < 8; ++r) { t0[r] = acc0[r]; t1[r] = acc1[r]; }
  store8(&s.ybT[laneLo * C_C + m0 + laneHi * 8], t0);
  store8(&s.ybT[(16 + laneLo) * C_C + m0 + laneHi * 8], t1);
}

// Channel GEMM: acc[o][v] = sum_c Wl[o][c] * B[v][c]  (M=128, N=32, K=128)
__device__ __forceinline__ void gemm_chan(const unsigned short* Wl,
                                          const unsigned short* B,
                                          f32x8& acc0, f32x8& acc1,
                                          int m0, int laneLo, int laneHi) {
  f32x8 a0 = {0, 0, 0, 0, 0, 0, 0, 0};
  f32x8 a1 = {0, 0, 0, 0, 0, 0, 0, 0};
#pragma unroll
  for (int kc = 0; kc < 4; ++kc) {
    bf16x16 af  = ld_frag(Wl, m0 + laneLo, C_C, kc * 32, laneHi);
    bf16x16 bf0 = ld_frag(B,  laneLo,      C_C, kc * 32, laneHi);
    bf16x16 bf1 = ld_frag(B,  16 + laneLo, C_C, kc * 32, laneHi);
    a0 = wmma_bf16(af, bf0, a0);
    a1 = wmma_bf16(af, bf1, a1);
  }
  acc0 = a0; acc1 = a1;
}

// xb[c][v] = bf16(z_t[c][v] + pe[c]); packed pair writes, zero padding v>=25.
__device__ __forceinline__ void build_xb(Smem& s, int tid) {
  unsigned int* xw = (unsigned int*)s.xb;
  for (int idx = tid; idx < C_C * 16; idx += 256) {
    int c = idx >> 4, v0 = (idx & 15) * 2;
    float pc = s.pec[c];
    float a = (v0 < V_V)     ? (s.zt[c * V_V + v0] + pc)     : 0.f;
    float b = (v0 + 1 < V_V) ? (s.zt[c * V_V + v0 + 1] + pc) : 0.f;
    xw[idx] = pkbf(a, b);
  }
}

// One ODE evaluation: input z_t (+pe) already in s.xb; writes f_t to `fout`.
__device__ __forceinline__ void ode_core(Smem& s, float* fout,
                                         int m0, int laneLo, int laneHi) {
#pragma unroll 1
  for (int l = 0; l < 3; ++l) {
    gemm_graph(s, m0, laneLo, laneHi);
    __syncthreads();
    f32x8 a0, a1;
    gemm_chan(s.W[l], s.ybT, a0, a1, m0, laneLo, laneHi);
    if (l < 2) {
      // tanh -> next x, row-major [c][v] (scattered b16 stores)
#pragma unroll
      for (int r = 0; r < 8; ++r) {
        int o = m0 + laneHi * 8 + r;
        float bo = s.bias[l][o];
        s.xb[o * 32 + laneLo]      = f2bf(fast_tanh(a0[r] + bo));
        s.xb[o * 32 + 16 + laneLo] = f2bf(fast_tanh(a1[r] + bo));
      }
    } else {
      // tanh -> x3 transposed [v][c] so the projection consumes it as B
      float t0[8], t1[8];
#pragma unroll
      for (int r = 0; r < 8; ++r) {
        float bo = s.bias[2][m0 + laneHi * 8 + r];
        t0[r] = fast_tanh(a0[r] + bo);
        t1[r] = fast_tanh(a1[r] + bo);
      }
      store8(&s.yb2[laneLo * C_C + m0 + laneHi * 8], t0);
      store8(&s.yb2[(16 + laneLo) * C_C + m0 + laneHi * 8], t1);
    }
    __syncthreads();
  }
  // final projection: f = Wp * x3 + bp (f32 result into ring slot)
  f32x8 a0, a1;
  gemm_chan(s.W[3], s.yb2, a0, a1, m0, laneLo, laneHi);
#pragma unroll
  for (int r = 0; r < 8; ++r) {
    int o = m0 + laneHi * 8 + r;
    float bo = s.bias[3][o];
    fout[o * V_V + laneLo] = a0[r] + bo;
    if (laneLo < V_V - 16) fout[o * V_V + 16 + laneLo] = a1[r] + bo;
  }
}

__global__ __launch_bounds__(256) void infogcn_ab6_kernel(
    const float* __restrict__ h, const float* __restrict__ mrand,
    const float* __restrict__ A, const float* __restrict__ pe,
    const float* __restrict__ W1, const float* __restrict__ b1,
    const float* __restrict__ W2, const float* __restrict__ b2,
    const float* __restrict__ W3, const float* __restrict__ b3,
    const float* __restrict__ Wp, const float* __restrict__ bp,
    float* __restrict__ out) {
  __shared__ Smem s;
  const int n = blockIdx.x;
  const int tid = threadIdx.x;
  const int lane = tid & 31;
  const int laneLo = lane & 15, laneHi = lane >> 4;
  const int m0 = (tid >> 5) * 16;

  const float* hn = h + (size_t)n * C_C * TV;
  float* on = out + (size_t)n * C_C * TV;

  // ---- one-time per block: convert weights/A to bf16 in LDS, biases f32 ----
  const float* Ws[4] = {W1, W2, W3, Wp};
  for (int l = 0; l < 4; ++l) {
    unsigned int* wd = (unsigned int*)s.W[l];
    const float* wsrc = Ws[l];
    for (int idx = tid; idx < C_C * C_C / 2; idx += 256)
      wd[idx] = pkbf(wsrc[idx * 2], wsrc[idx * 2 + 1]);
  }
  for (int idx = tid; idx < 32 * 32; idx += 256) {
    int v = idx >> 5, u = idx & 31;
    s.Ab[idx] = f2bf((v < V_V && u < V_V) ? A[v * V_V + u] : 0.f);
  }
  const float* Bs[4] = {b1, b2, b3, bp};
  if (tid < C_C)
    for (int l = 0; l < 4; ++l) s.bias[l][tid] = Bs[l][tid];

  // out[:, :, 0:6, :] = h[:, :, 0:6, :]
  for (int idx = tid; idx < C_C * 6 * V_V; idx += 256) {
    int c = idx / (6 * V_V), r = idx % (6 * V_V);
    on[c * TV + r] = hn[c * TV + r];
  }
  __syncthreads();

  const float coef[6] = {4277.f / 1440.f, -7923.f / 1440.f, 9982.f / 1440.f,
                         -7298.f / 1440.f, 2877.f / 1440.f, -475.f / 1440.f};

  // ---- initial f_k = ode(k, h[:,:,k,:]) for k=0..5, ring[5-k] = f_k ----
  for (int k = 0; k < 6; ++k) {
    for (int idx = tid; idx < C_C * V_V; idx += 256) {
      int c = idx / V_V, v = idx - c * V_V;
      float z = hn[c * TV + k * V_V + v];
      s.zt[idx] = z;
      if (k == 5) s.zc[idx] = z;  // z_cand = h[:,:,5]
    }
    if (tid < C_C) s.pec[tid] = pe[tid * (3 * T_T) + 3 * k];
    __syncthreads();
    build_xb(s, tid);
    __syncthreads();
    ode_core(s, s.ring[5 - k], m0, laneLo, laneHi);
    __syncthreads();
  }

  // prefetch h[:,:,5,:] for the first scan step
  issue_hslice_async(s, hn, 5, tid);

  // ---- AB6 scan: t = 5..62, output time t+1 ----
  int head = 0;
  for (int i = 0; i < T_T - 6; ++i) {
    const int t = 5 + i;
    const bool m = mrand[i * N_S + n] < 0.8f;  // uniform across block

    // retire the staged slice for this step (per-wave), then use it
    wait_async0();
    for (int idx = tid; idx < C_C * V_V; idx += 256)
      s.zt[idx] = m ? s.zc[idx] : s.hstage[idx];
    if (tid < C_C) s.pec[tid] = pe[tid * (3 * T_T) + 3 * t];
    __syncthreads();

    // hstage is free now: overlap next slice's load with the whole ODE chain
    if (i + 1 < T_T - 6) issue_hslice_async(s, hn, t + 1, tid);

    // phase B: z_next = z_t + sum_k coef[k]*ring[k]; also build xb = z_t + pe
    const float* rp[6];
#pragma unroll
    for (int k2 = 0; k2 < 6; ++k2) rp[k2] = s.ring[(head + k2) % 6];
    for (int idx = tid; idx < C_C * V_V; idx += 256) {
      float acc = s.zt[idx];
#pragma unroll
      for (int k2 = 0; k2 < 6; ++k2) acc += coef[k2] * rp[k2][idx];
      s.zc[idx] = acc;  // z_cand for next step
      int c = idx / V_V, v = idx - c * V_V;
      on[c * TV + (t + 1) * V_V + v] = acc;
    }
    build_xb(s, tid);
    __syncthreads();

    // phase C: f_t = ode(t, z_t) into new ring head (evicted oldest slot)
    head = (head + 5) % 6;
    ode_core(s, s.ring[head], m0, laneLo, laneHi);
    __syncthreads();
  }
}

extern "C" void kernel_launch(void* const* d_in, const int* in_sizes, int n_in,
                              void* d_out, int out_size, void* d_ws,
                              size_t ws_size, hipStream_t stream) {
  (void)in_sizes; (void)n_in; (void)d_ws; (void)ws_size; (void)out_size;
  const float* h  = (const float*)d_in[0];
  const float* mr = (const float*)d_in[1];
  const float* A  = (const float*)d_in[2];
  const float* pe = (const float*)d_in[3];
  const float* W1 = (const float*)d_in[4];
  const float* b1 = (const float*)d_in[5];
  const float* W2 = (const float*)d_in[6];
  const float* b2 = (const float*)d_in[7];
  const float* W3 = (const float*)d_in[8];
  const float* b3 = (const float*)d_in[9];
  const float* Wp = (const float*)d_in[10];
  const float* bp = (const float*)d_in[11];
  float* out = (float*)d_out;
  infogcn_ab6_kernel<<<dim3(N_S), dim3(256), 0, stream>>>(
      h, mr, A, pe, W1, b1, W2, b2, W3, b3, Wp, bp, out);
}
